// Attention_21474836480706
// MI455X (gfx1250) — matrix operand order repaired and checked by reading, not compiled, not run on hardware
//
#include <hip/hip_runtime.h>
#include <hip/hip_bf16.h>
#include <stdint.h>

// ---------------------------------------------------------------------------
// MI455X (gfx1250) attention: bf16 WMMA (f32 accum) everywhere.
// GEMMs: 128x64 workgroup tiles, BK=64 double-buffered async global->LDS,
// A-tiles stored in LDS pre-swizzled into WMMA fragment order (contig b128s).
// B=4 N=1024 H=16 D=72 HID=1152 ATT=1152
// ---------------------------------------------------------------------------

typedef __attribute__((ext_vector_type(16))) __bf16 v16bf;
typedef __attribute__((ext_vector_type(8)))  float  v8f;
typedef __bf16 bf16_t;

#define B_      4
#define N_      1024
#define H_      16
#define D_      72
#define HID_    1152
#define ATT_    1152
#define QKV3_   3456
#define DPADQK  96      // K-dim padding for S = Q@K^T (multiple of 32)
#define DPADV   80      // d padding for O = P@V (multiple of 16)
#define M_      (B_*N_) // 4096

// GEMM block tiling
#define BM 128
#define BN 64
#define BK 64           // LDS stage depth (two WMMA k-steps)
#define LDK 72          // padded LDS row (elements) -> 144B; spreads b128 reads
                        // across all 16 four-bank groups

union FragBF { v16bf v; uint32_t u[8]; };

__device__ inline v8f zero8() {
  v8f z = {0.f,0.f,0.f,0.f,0.f,0.f,0.f,0.f};
  return z;
}

__device__ inline v8f wmma_bf16(v16bf a, v16bf b, v8f c) {
  // (neg_a, A, neg_b, B, c_mod, C, reuse_a, reuse_b)
  return __builtin_amdgcn_wmma_f32_16x16x32_bf16(false, a, false, b, (short)0, c, false, false);
}

// A-matrix 16x32 bf16 fragment from row-major GLOBAL storage (K contiguous).
// ISA: lanes 0-15: M=lane; VGPR j<4 -> K=2j,2j+1 ; j>=4 -> K=16+2(j-4)..
//      lanes 16-31: same with K offset +8.
__device__ inline v16bf load_a_frag(const bf16_t* __restrict__ base, int ldk,
                                    int row0, int kbase, int lane) {
  FragBF f;
  const int b32 = (lane & 16) ? 4 : 0;
  const uint32_t* rp = (const uint32_t*)(base + (size_t)(row0 + (lane & 15)) * ldk + kbase);
  #pragma unroll
  for (int j = 0; j < 4; ++j) f.u[j] = rp[b32 + j];
  #pragma unroll
  for (int j = 4; j < 8; ++j) f.u[j] = rp[4 + b32 + j];
  return f.v;
}

// A fragment from LDS tile that was stored pre-swizzled into fragment order:
// per 64B half-row, quarters permuted [q0 q1 q2 q3] -> [q0 q2 q1 q3], so each
// lane's 16 elements are 32B contiguous.
__device__ inline v16bf load_a_frag_sw(const bf16_t* __restrict__ base,
                                       int row0, int kbase, int lane) {
  FragBF f;
  const uint32_t* rp = (const uint32_t*)(base + (size_t)(row0 + (lane & 15)) * LDK
                                         + kbase + ((lane & 16) ? 16 : 0));
  #pragma unroll
  for (int j = 0; j < 8; ++j) f.u[j] = rp[j];
  return f.v;
}

// B-matrix 32x16 bf16 fragment from column-major storage baseT[col*ldk + k].
// ISA: lanes 0-15: N=lane; VGPR j<4 -> K=2j,2j+1 ; j>=4 -> K=16+2(j-4).. ; lanes 16-31: K offset +8.
__device__ inline v16bf load_b_frag(const bf16_t* __restrict__ baseT, int ldk,
                                    int col0, int kbase, int lane) {
  FragBF f;
  const int koff = (lane & 16) ? 4 : 0;
  const uint32_t* cp = (const uint32_t*)(baseT + (size_t)(col0 + (lane & 15)) * ldk + kbase);
  #pragma unroll
  for (int j = 0; j < 4; ++j) f.u[j] = cp[koff + j];
  #pragma unroll
  for (int j = 4; j < 8; ++j) f.u[j] = cp[4 + koff + j];
  return f.v;
}

// async 16B global -> LDS copy (per lane), tracked by ASYNCcnt
__device__ inline void async_copy_b128(uint32_t lds_addr, const void* gptr) {
  uint64_t ga = (uint64_t)(uintptr_t)gptr;
  asm volatile("global_load_async_to_lds_b128 %0, %1, off"
               :: "v"(lds_addr), "v"(ga) : "memory");
}
__device__ inline void wait_async0() {
  asm volatile("s_wait_asynccnt 0x0" ::: "memory");
}

// ---------------------------------------------------------------------------
// Conversion kernels
// ---------------------------------------------------------------------------
__global__ void cvt_kernel(const float* __restrict__ src, bf16_t* __restrict__ dst, int n) {
  int i = blockIdx.x * blockDim.x + threadIdx.x;
  if (i < n) dst[i] = (bf16_t)src[i];
}

// dst[c][r] = (bf16) src[r][c], R,C multiples of 32
__global__ __launch_bounds__(256) void transpose_cvt_kernel(const float* __restrict__ src,
                                                            bf16_t* __restrict__ dst,
                                                            int R, int C) {
  __shared__ float tile[32][33];
  const int tx = threadIdx.x & 31;
  const int ty = threadIdx.x >> 5;
  const int c0 = blockIdx.x * 32;
  const int r0 = blockIdx.y * 32;
  #pragma unroll
  for (int i = ty; i < 32; i += 8)
    tile[i][tx] = src[(size_t)(r0 + i) * C + (c0 + tx)];
  __syncthreads();
  #pragma unroll
  for (int i = ty; i < 32; i += 8)
    dst[(size_t)(c0 + i) * R + (r0 + tx)] = (bf16_t)tile[tx][i];
}

// zero the d=72..95 pad columns of qpad/kpad each call (ws is not re-zeroed)
__global__ void zeropad_kernel(bf16_t* __restrict__ qpad, bf16_t* __restrict__ kpad) {
  const int padw = DPADQK - D_;
  int i = blockIdx.x * blockDim.x + threadIdx.x;
  int total = B_ * H_ * N_ * padw;
  if (i < total) {
    int row = i / padw;
    int d   = D_ + i % padw;
    qpad[(size_t)row * DPADQK + d] = (bf16_t)0.f;
    kpad[(size_t)row * DPADQK + d] = (bf16_t)0.f;
  }
}

// ---------------------------------------------------------------------------
// Shared GEMM core: 256 threads, BMxBN tile, BK=64 async double-buffered LDS.
// A row-major [M][K] (stored swizzled in LDS), B column-major [Ncol][K].
// ---------------------------------------------------------------------------
#define GEMM_CORE(Aptr, Bptr, KTOT)                                              \
  __shared__ __align__(16) bf16_t shA[2][BM * LDK];                              \
  __shared__ __align__(16) bf16_t shB[2][BN * LDK];                              \
  const int tid  = threadIdx.x;                                                  \
  const int lane = tid & 31;                                                     \
  const int wave = tid >> 5;                                                     \
  const int wm   = (wave & 3) * 32;                                              \
  const int wn   = (wave >> 2) * 32;                                             \
  const int mb   = blockIdx.x % (M_ / BM);                                       \
  const int nb   = blockIdx.x / (M_ / BM);                                       \
  const int m0g  = mb * BM;                                                      \
  const int n0g  = nb * BN;                                                      \
  auto prefetch = [&](int buf, int k) {                                          \
    _Pragma("unroll")                                                            \
    for (int i = 0; i < 4; ++i) {         /* A: 128 rows x 8 16B-chunks */       \
      int idx = tid + i * 256;            /* 0..1023 */                          \
      int row = idx >> 3;                 /* 0..127 */                           \
      int g   = idx & 7;                  /* chunk in row */                     \
      int grp = g >> 2;                   /* 64B half-row */                     \
      int q   = g & 3;                    /* quarter */                          \
      int qp  = ((q & 1) << 1) | (q >> 1);/* 0->0,1->2,2->1,3->3 swizzle */      \
      async_copy_b128((uint32_t)(uintptr_t)&shA[buf][row * LDK]                  \
                          + grp * 64 + qp * 16,                                  \
                      (const char*)(Aptr + (size_t)(m0g + row) * (KTOT) + k)     \
                          + g * 16);                                             \
    }                                                                            \
    _Pragma("unroll")                                                            \
    for (int i = 0; i < 2; ++i) {         /* B: 64 rows x 8 16B-chunks */        \
      int idx = tid + i * 256;            /* 0..511 */                           \
      int row = idx >> 3;                 /* 0..63 */                            \
      int qo  = (idx & 7) * 16;                                                  \
      async_copy_b128((uint32_t)(uintptr_t)&shB[buf][row * LDK] + qo,            \
                      (const char*)(Bptr + (size_t)(n0g + row) * (KTOT) + k) + qo);\
    }                                                                            \
  };                                                                             \
  v8f c00 = zero8(), c01 = zero8(), c10 = zero8(), c11 = zero8();                \
  prefetch(0, 0);                                                                \
  const int KT = (KTOT) / BK;                                                    \
  for (int kt = 0; kt < KT; ++kt) {                                              \
    wait_async0();                                                               \
    __syncthreads();                                                             \
    if (kt + 1 < KT) prefetch((kt + 1) & 1, (kt + 1) * BK);                      \
    const bf16_t* At = shA[kt & 1];                                              \
    const bf16_t* Bt = shB[kt & 1];                                              \
    _Pragma("unroll")                                                            \
    for (int ks = 0; ks < 2; ++ks) {      /* two 16x16x32 k-steps per stage */   \
      const int kb = ks * 32;                                                    \
      v16bf a0 = load_a_frag_sw(At, wm,      kb, lane);                          \
      v16bf a1 = load_a_frag_sw(At, wm + 16, kb, lane);                          \
      v16bf b0 = load_b_frag(Bt, LDK, wn,      kb, lane);                        \
      v16bf b1 = load_b_frag(Bt, LDK, wn + 16, kb, lane);                        \
      c00 = wmma_bf16(a0, b0, c00);                                              \
      c01 = wmma_bf16(a0, b1, c01);                                              \
      c10 = wmma_bf16(a1, b0, c10);                                              \
      c11 = wmma_bf16(a1, b1, c11);                                              \
    }                                                                            \
    __syncthreads();                                                             \
  }                                                                              \
  const int m0 = m0g + wm;                                                       \
  const int n0 = n0g + wn;

// ---------------------------------------------------------------------------
// QKV GEMM: C[4096][3456] = xbf @ wqkv + b ; scatter into qpad/kpad/vT
// ---------------------------------------------------------------------------
__device__ inline void scatter_qkv(v8f acc, int m0, int col, int lane,
                                   const float* __restrict__ bias,
                                   bf16_t* __restrict__ qpad,
                                   bf16_t* __restrict__ kpad,
                                   bf16_t* __restrict__ vT) {
  const float bv = bias[col];
  const int which = col / ATT_;
  const int cc    = col % ATT_;
  const int h     = cc / D_;
  const int d     = cc % D_;
  #pragma unroll
  for (int r = 0; r < 8; ++r) {
    const int row = m0 + ((lane & 16) ? 8 + r : r);
    const int bb  = row >> 10;
    const int n   = row & (N_ - 1);
    const float val = acc[r] + bv;
    const int bh = bb * H_ + h;
    if (which == 0)      qpad[((size_t)bh * N_ + n) * DPADQK + d] = (bf16_t)val;
    else if (which == 1) kpad[((size_t)bh * N_ + n) * DPADQK + d] = (bf16_t)val;
    else                 vT  [((size_t)bh * DPADV + d) * N_ + n]  = (bf16_t)val;
  }
}

__global__ __launch_bounds__(256) void qkv_gemm_kernel(const bf16_t* __restrict__ xbf,
                                                       const bf16_t* __restrict__ wT,
                                                       const float*  __restrict__ bias,
                                                       bf16_t* __restrict__ qpad,
                                                       bf16_t* __restrict__ kpad,
                                                       bf16_t* __restrict__ vT) {
  GEMM_CORE(xbf, wT, HID_)
  const int cl = lane & 15;
  scatter_qkv(c00, m0,      n0      + cl, lane, bias, qpad, kpad, vT);
  scatter_qkv(c01, m0,      n0 + 16 + cl, lane, bias, qpad, kpad, vT);
  scatter_qkv(c10, m0 + 16, n0      + cl, lane, bias, qpad, kpad, vT);
  scatter_qkv(c11, m0 + 16, n0 + 16 + cl, lane, bias, qpad, kpad, vT);
}

// ---------------------------------------------------------------------------
// Output projection: out[4096][1152] = attOut @ w_out + b_out (f32 out)
// ---------------------------------------------------------------------------
__global__ __launch_bounds__(256) void out_gemm_kernel(const bf16_t* __restrict__ abf,
                                                       const bf16_t* __restrict__ wT2,
                                                       const float*  __restrict__ bias,
                                                       float* __restrict__ out) {
  GEMM_CORE(abf, wT2, ATT_)
  const int cl = lane & 15;
  #pragma unroll
  for (int s = 0; s < 4; ++s) {
    const v8f acc = (s == 0) ? c00 : (s == 1) ? c01 : (s == 2) ? c10 : c11;
    const int mm  = m0 + ((s >= 2) ? 16 : 0);
    const int col = n0 + ((s & 1) ? 16 : 0) + cl;
    const float bv = bias[col];
    #pragma unroll
    for (int r = 0; r < 8; ++r) {
      const int row = mm + ((lane & 16) ? 8 + r : r);
      out[(size_t)row * HID_ + col] = acc[r] + bv;
    }
  }
}

// ---------------------------------------------------------------------------
// Flash attention: one wave per (b,h, 16-query block). Online softmax.
// ---------------------------------------------------------------------------
__global__ __launch_bounds__(32) void attn_kernel(const bf16_t* __restrict__ qpad,
                                                  const bf16_t* __restrict__ kpad,
                                                  const bf16_t* __restrict__ vT,
                                                  bf16_t* __restrict__ attOut) {
  // P staging, stored directly in fragment order:
  // per row: [cols 0-7 | cols 16-23 | cols 8-15 | cols 24-31]
  __shared__ __align__(16) bf16_t plds[16][32];
  const int lane = threadIdx.x & 31;
  const int qb = blockIdx.x & 63;
  const int bh = blockIdx.x >> 6;
  const bf16_t* qbase = qpad + (size_t)bh * N_ * DPADQK;
  const bf16_t* kbase = kpad + (size_t)bh * N_ * DPADQK;
  const bf16_t* vbase = vT   + (size_t)bh * DPADV * N_;
  const float scale = 0.11785113019775793f;  // 1/sqrt(72)

  v16bf aq[3];
  #pragma unroll
  for (int t = 0; t < 3; ++t) aq[t] = load_a_frag(qbase, DPADQK, qb * 16, t * 32, lane);

  float mrow[8], lrow[8];
  #pragma unroll
  for (int r = 0; r < 8; ++r) { mrow[r] = -3.0e38f; lrow[r] = 0.f; }
  v8f o[5];
  #pragma unroll
  for (int f = 0; f < 5; ++f) o[f] = zero8();

  const int c = lane & 15;
  const int pos0 = (c < 8) ? c : c + 8;          // col c       -> frag pos
  const int pos1 = (c < 8) ? c + 8 : c + 16;     // col 16 + c  -> frag pos
  const int prow = (lane & 16) ? 8 : 0;

  for (int kc = 0; kc < N_ / 32; ++kc) {
    v8f s0 = zero8(), s1 = zero8();
    #pragma unroll
    for (int t = 0; t < 3; ++t) {
      v16bf bk0 = load_b_frag(kbase, DPADQK, kc * 32,      t * 32, lane);
      v16bf bk1 = load_b_frag(kbase, DPADQK, kc * 32 + 16, t * 32, lane);
      s0 = wmma_bf16(aq[t], bk0, s0);
      s1 = wmma_bf16(aq[t], bk1, s1);
    }
    float p0[8], p1[8];
    #pragma unroll
    for (int r = 0; r < 8; ++r) {
      float v0 = s0[r] * scale, v1 = s1[r] * scale;
      float tm = fmaxf(v0, v1);
      #pragma unroll
      for (int off = 1; off < 16; off <<= 1) tm = fmaxf(tm, __shfl_xor(tm, off, 16));
      float newm  = fmaxf(mrow[r], tm);
      float alpha = __expf(mrow[r] - newm);
      p0[r] = __expf(v0 - newm);
      p1[r] = __expf(v1 - newm);
      float rs = p0[r] + p1[r];
      #pragma unroll
      for (int off = 1; off < 16; off <<= 1) rs += __shfl_xor(rs, off, 16);
      lrow[r] = lrow[r] * alpha + rs;
      mrow[r] = newm;
      #pragma unroll
      for (int f = 0; f < 5; ++f) o[f][r] *= alpha;
    }
    // P -> LDS in fragment order. Single wave: DS ops in-order.
    #pragma unroll
    for (int r = 0; r < 8; ++r) {
      plds[prow + r][pos0] = (bf16_t)p0[r];
      plds[prow + r][pos1] = (bf16_t)p1[r];
    }
    // contiguous 32B fragment read -> 2x ds_load_b128
    v16bf ap;
    {
      FragBF f;
      const uint32_t* rp = (const uint32_t*)&plds[lane & 15][0] + ((lane & 16) ? 8 : 0);
      #pragma unroll
      for (int j = 0; j < 8; ++j) f.u[j] = rp[j];
      ap = f.v;
    }
    #pragma unroll
    for (int df = 0; df < 5; ++df) {
      v16bf bv = load_b_frag(vbase, N_, df * 16, kc * 32, lane);
      o[df] = wmma_bf16(ap, bv, o[df]);
    }
  }
  const int h  = bh % H_;
  const int bb = bh / H_;
  #pragma unroll
  for (int r = 0; r < 8; ++r) {
    const int q = qb * 16 + ((lane & 16) ? 8 + r : r);
    const float inv = 1.0f / lrow[r];
    #pragma unroll
    for (int df = 0; df < 5; ++df) {
      const int d = df * 16 + (lane & 15);
      if (d < D_)
        attOut[(size_t)(bb * N_ + q) * ATT_ + h * D_ + d] = (bf16_t)(o[df][r] * inv);
    }
  }
}

// ---------------------------------------------------------------------------
extern "C" void kernel_launch(void* const* d_in, const int* in_sizes, int n_in,
                              void* d_out, int out_size, void* d_ws, size_t ws_size,
                              hipStream_t stream) {
  const float* x     = (const float*)d_in[0];
  const float* w_qkv = (const float*)d_in[1];
  const float* b_qkv = (const float*)d_in[2];
  const float* w_out = (const float*)d_in[3];
  const float* b_out = (const float*)d_in[4];
  float* out = (float*)d_out;

  char* ws = (char*)d_ws;
  bf16_t* xbf    = (bf16_t*)(ws);                 // 4096*1152*2  =  9,437,184
  bf16_t* wqkvT  = (bf16_t*)(ws +  9437184);      // 3456*1152*2  =  7,962,624
  bf16_t* woutT  = (bf16_t*)(ws + 17399808);      // 1152*1152*2  =  2,654,208
  bf16_t* qpad   = (bf16_t*)(ws + 20054016);      // 64*1024*96*2 = 12,582,912
  bf16_t* kpad   = (bf16_t*)(ws + 32636928);      // same         = 12,582,912
  bf16_t* vT     = (bf16_t*)(ws + 45219840);      // 64*80*1024*2 = 10,485,760
  bf16_t* attOut = (bf16_t*)(ws + 55705600);      // 4096*1152*2  =  9,437,184
  // total ~65.1 MB of workspace

  // 1) precision conversion / weight transposition
  {
    int n = M_ * HID_;
    cvt_kernel<<<(n + 255) / 256, 256, 0, stream>>>(x, xbf, n);
  }
  transpose_cvt_kernel<<<dim3(QKV3_ / 32, HID_ / 32), 256, 0, stream>>>(w_qkv, wqkvT, HID_, QKV3_);
  transpose_cvt_kernel<<<dim3(HID_  / 32, ATT_ / 32), 256, 0, stream>>>(w_out, woutT, ATT_, HID_);
  {
    int n = B_ * H_ * N_ * (DPADQK - D_);
    zeropad_kernel<<<(n + 255) / 256, 256, 0, stream>>>(qpad, kpad);
  }

  // 2) QKV projection (WMMA + async LDS staging), scatter into attention layouts
  qkv_gemm_kernel<<<(M_ / BM) * (QKV3_ / BN), 256, 0, stream>>>(xbf, wqkvT, b_qkv,
                                                                qpad, kpad, vT);
  // 3) flash attention (WMMA + online softmax)
  attn_kernel<<<B_ * H_ * (N_ / 16), 32, 0, stream>>>(qpad, kpad, vT, attOut);

  // 4) output projection (WMMA + async LDS staging)
  out_gemm_kernel<<<(M_ / BM) * (HID_ / BN), 256, 0, stream>>>(attOut, woutT, b_out, out);
}
